// MoELayer_84370337563092
// MI455X (gfx1250) — matrix operand order repaired and checked by reading, not compiled, not run on hardware
//
#include <hip/hip_runtime.h>
#include <hip/hip_bf16.h>

#define DM 1024   // d_model
#define DF 4096   // d_ff
#define NE 8      // experts
#define TMT 64    // tokens per tile
#define KC 32     // ff chunk
#define HP 40     // LDS pitch (bf16 elems) for h tile
#define XP 1032   // LDS pitch (bf16 elems) for x tile (16B-aligned rows, bank skew)
#define W2P 40    // LDS pitch (bf16 elems) for TDM-staged W2 chunk (64B data + 16B pad)

typedef __bf16 v16bf __attribute__((ext_vector_type(16)));
typedef __bf16 v8bf  __attribute__((ext_vector_type(8)));
typedef float  v8f   __attribute__((ext_vector_type(8)));
typedef unsigned int u32x4 __attribute__((ext_vector_type(4)));
typedef int i32x4 __attribute__((ext_vector_type(4)));
typedef int i32x8 __attribute__((ext_vector_type(8)));

// LDS layout (dynamic shared):
#define W2S_BYTES (DM * W2P * 2)            // 81920
#define XS_BYTES  (TMT * XP * 2)            // 132096
#define HS_BYTES  (TMT * HP * 2)            // 5120
#define SMEM_BYTES (W2S_BYTES + XS_BYTES + HS_BYTES + 256 + 256)

// A-matrix fragment (16x32 bf16): lane = 16*half + (M&15).
// elems 0..7 <-> K = half*8 + j ; elems 8..15 <-> K = 16 + half*8 + j
static __device__ inline v16bf frag_a(const __bf16* p) {
  v8bf lo = *reinterpret_cast<const v8bf*>(p);
  v8bf hi = *reinterpret_cast<const v8bf*>(p + 16);
  v16bf r;
#pragma unroll
  for (int i = 0; i < 8; ++i) { r[i] = lo[i]; r[i + 8] = hi[i]; }
  return r;
}

// B-matrix fragment (32x16 bf16): lane = 16*half + N;
// elems 0..7 <-> K = half*8 + j ; elems 8..15 <-> K = 16 + half*8 + j
static __device__ inline v16bf frag_b(const __bf16* p) {
  v8bf lo = *reinterpret_cast<const v8bf*>(p);
  v8bf hi = *reinterpret_cast<const v8bf*>(p + 16);
  v16bf r;
#pragma unroll
  for (int i = 0; i < 8; ++i) { r[i] = lo[i]; r[i + 8] = hi[i]; }
  return r;
}

// ---------------- elementwise fp32 -> bf16 ----------------
__global__ void k_cvt_bf16(const float* __restrict__ in, __bf16* __restrict__ out, int n) {
  int i = blockIdx.x * blockDim.x + threadIdx.x;
  if (i < n) out[i] = (__bf16)in[i];
}

// -------- tiled transpose+convert: in [E][R][C] f32 -> out [E][C][R] bf16 --------
__global__ void k_transpose_bf16(const float* __restrict__ in, __bf16* __restrict__ out,
                                 int R, int C) {
  __shared__ float tile[32][33];
  const size_t eoff = (size_t)blockIdx.z * R * C;
  const int c0 = blockIdx.x * 32, r0 = blockIdx.y * 32;
  const int tx = threadIdx.x, ty = threadIdx.y;   // (32, 8)
#pragma unroll
  for (int j = 0; j < 4; ++j)
    tile[ty + 8 * j][tx] = in[eoff + (size_t)(r0 + ty + 8 * j) * C + c0 + tx];
  __syncthreads();
#pragma unroll
  for (int j = 0; j < 4; ++j)
    out[eoff + (size_t)(c0 + ty + 8 * j) * R + r0 + tx] = (__bf16)tile[tx][ty + 8 * j];
}

// ---------------- gating: logits, top-2, softmax, routing lists ----------------
__global__ void k_gate(const float* __restrict__ x, const float* __restrict__ Wg,
                       const float* __restrict__ bg, int* __restrict__ cnt,
                       int* __restrict__ list, float* __restrict__ wlist, int T) {
  const int wid = threadIdx.x >> 5, lane = threadIdx.x & 31;
  const int t = blockIdx.x * 8 + wid;
  if (t >= T) return;
  float p[NE];
#pragma unroll
  for (int e = 0; e < NE; ++e) p[e] = 0.f;
  const float* xr = x + (size_t)t * DM;
  for (int k = lane; k < DM; k += 32) {
    const float xv = xr[k];
    const float* wr = Wg + k * NE;
#pragma unroll
    for (int e = 0; e < NE; ++e) p[e] += xv * wr[e];
  }
#pragma unroll
  for (int off = 16; off > 0; off >>= 1)
#pragma unroll
    for (int e = 0; e < NE; ++e) p[e] += __shfl_xor(p[e], off, 32);
  if (lane == 0) {
    float g[NE];
#pragma unroll
    for (int e = 0; e < NE; ++e) g[e] = p[e] + bg[e];
    int e1 = 0; float g1 = g[0];
#pragma unroll
    for (int e = 1; e < NE; ++e) if (g[e] > g1) { g1 = g[e]; e1 = e; }
    int e2 = -1; float g2 = -1e30f;
#pragma unroll
    for (int e = 0; e < NE; ++e) if (e != e1 && g[e] > g2) { g2 = g[e]; e2 = e; }
    float w2 = __expf(g2 - g1);
    const float s = 1.f + w2;
    const float w1 = 1.f / s; w2 /= s;
    int pos = atomicAdd(&cnt[e1], 1);
    list[e1 * T + pos] = t; wlist[e1 * T + pos] = w1;
    pos = atomicAdd(&cnt[e2], 1);
    list[e2 * T + pos] = t; wlist[e2 * T + pos] = w2;
  }
}

// Build D# and issue TDM load of W2 chunk [1024 rows x 32 bf16] -> LDS (pitch 80B).
static __device__ inline void tdm_load_w2_chunk(const __bf16* gsrc, unsigned int lds_off) {
  const unsigned long long ga = (unsigned long long)(size_t)gsrc;
  u32x4 g0;
  g0[0] = 1u;                                   // count=1, no gather
  g0[1] = lds_off;                              // lds_addr (bytes)
  g0[2] = (unsigned int)ga;                     // global_addr[31:0]
  g0[3] = ((unsigned int)(ga >> 32) & 0x01FFFFFFu) | 0x80000000u;  // addr[56:32] | type=2
  i32x8 g1;
  g1[0] = (1 << 16)        // data_size = 2 bytes
        | (1 << 20)        // pad_enable
        | (3 << 22)        // pad_interval: every 16 DWORDs (64B)
        | (3 << 25);       // pad_amount: 4 DWORDs (16B)  -> LDS pitch 80B
  g1[1] = (DF & 0xFFFF) << 16;     // tensor_dim0 = 4096 (low16 at bits 63:48)
  g1[2] = (8192 & 0xFFFF) << 16;   // tensor_dim0 hi=0 | tensor_dim1 = 8192
  g1[3] = (KC & 0xFFFF) << 16;     // tensor_dim1 hi=0 | tile_dim0 = 32
  g1[4] = DM;                      // tile_dim1 = 1024 | tile_dim2 = 0
  g1[5] = DF;                      // tensor_dim0_stride = 4096 (elements)
  g1[6] = 0;                       // stride0 hi | tensor_dim1_stride lo
  g1[7] = 0;
  i32x4 z4;
  z4[0] = 0; z4[1] = 0; z4[2] = 0; z4[3] = 0;
  i32x8 z8;
#pragma unroll
  for (int i = 0; i < 8; ++i) z8[i] = 0;
  // 6-arg toolchain variant: (g0, g1, g2, g3, g4, cpol)
  __builtin_amdgcn_tensor_load_to_lds(g0, g1, z4, z4, z8, 0);
}

// ------- fused expert FFN: relu(x W1 + b1) W2 + b2, weighted combine -------
__global__ __launch_bounds__(256)
void k_moe_ffn(const __bf16* __restrict__ xb,    // [T][DM] bf16
               const __bf16* __restrict__ w1T,   // [E][DF][DM]  (K-major B frags)
               const __bf16* __restrict__ w2T,   // [E][DM][DF]  (K-major B frags)
               const float* __restrict__ b1,     // [E][DF]
               const float* __restrict__ b2,     // [E][DM]
               const int* __restrict__ cnt,
               const int* __restrict__ list,
               const float* __restrict__ wlist,
               float* __restrict__ out, int T) {
  extern __shared__ char smem[];
  __bf16* w2s = (__bf16*)smem;                               // [1024][W2P]
  __bf16* xs  = (__bf16*)(smem + W2S_BYTES);                 // [64][XP]
  __bf16* hs  = (__bf16*)(smem + W2S_BYTES + XS_BYTES);      // [64][HP]
  int*   toks = (int*)  (smem + W2S_BYTES + XS_BYTES + HS_BYTES);
  float* wts  = (float*)(smem + W2S_BYTES + XS_BYTES + HS_BYTES + 256);

  const int e = blockIdx.y;
  const int base = blockIdx.x * TMT;
  const int ne = cnt[e];
  if (base >= ne) return;
  const int rows = min(TMT, ne - base);

  const int tid = threadIdx.x;
  if (tid < TMT) {
    const bool ok = tid < rows;
    toks[tid] = ok ? list[e * T + base + tid] : 0;
    wts[tid]  = ok ? wlist[e * T + base + tid] : 0.f;
  }
  __syncthreads();

  // ---- stage gathered x tile into LDS once (read L2-resident bf16 x) ----
  for (int idx = tid; idx < TMT * (DM / 8); idx += 256) {
    const int r = idx >> 7, c = idx & 127;     // 128 8-elem chunks per row
    const v8bf v = *reinterpret_cast<const v8bf*>(xb + (size_t)toks[r] * DM + c * 8);
    *reinterpret_cast<v8bf*>(xs + r * XP + c * 8) = v;
  }

  const int wid = tid >> 5, lane = tid & 31;
  const int l15 = lane & 15, half = lane >> 4;
  const int mt = wid & 3, nt = wid >> 2;       // this wave's h tile (4x2 grid)

  const __bf16* aBase = xs + (mt * 16 + l15) * XP + half * 8;
  const unsigned int w2s_off = (unsigned int)(size_t)w2s;  // LDS byte offset

  v8f acc[4][8];
#pragma unroll
  for (int i = 0; i < 4; ++i)
#pragma unroll
    for (int j = 0; j < 8; ++j)
#pragma unroll
      for (int k = 0; k < 8; ++k) acc[i][j][k] = 0.f;

  __syncthreads();   // xs staged

  for (int ff0 = 0; ff0 < DF; ff0 += KC) {
    if (ff0) __syncthreads();    // all waves done with hs & w2s of prev chunk

    // async: TDM pulls this chunk's W2 [32 x 1024] stripe into LDS while GEMM1 runs
    if (wid == 0) {
      tdm_load_w2_chunk(w2T + (size_t)e * DM * DF + ff0, w2s_off);
    }
    if (ff0 + KC < DF)
      __builtin_prefetch(w1T + ((size_t)e * DF + ff0 + KC + nt * 16 + l15) * DM, 0, 1);

    // ---- GEMM1: h[64][32] = x[64][1024] @ W1[:, ff0:ff0+32]; one 16x16 tile/wave
    v8f hacc;
#pragma unroll
    for (int k = 0; k < 8; ++k) hacc[k] = 0.f;
    const __bf16* bBase =
        w1T + ((size_t)e * DF + ff0 + nt * 16 + l15) * DM + half * 8;
    for (int k0 = 0; k0 < DM; k0 += KC) {
      v16bf a = frag_a(aBase + k0);   // LDS
      v16bf b = frag_b(bBase + k0);   // global (L2-resident)
      hacc = __builtin_amdgcn_wmma_f32_16x16x32_bf16(
          false, a, false, b, (short)0, hacc, false, false);
    }

    // bias + ReLU, stage h for all waves (C layout: M = r + 8*half, N = l15)
    const float b1v = b1[e * DF + ff0 + nt * 16 + l15];
#pragma unroll
    for (int r = 0; r < 8; ++r) {
      const int row = mt * 16 + r + half * 8;
      float v = hacc[r] + b1v;
      v = v > 0.f ? v : 0.f;
      hs[row * HP + nt * 16 + l15] = (__bf16)v;
    }
    if (wid == 0) __builtin_amdgcn_s_wait_tensorcnt(0);   // TDM done (wave 0 issued)
    __syncthreads();                                      // hs + w2s visible to all

    // ---- GEMM2: y[64][1024] += h[64][32] @ W2[ff0:ff0+32][:]; wave owns 128 cols
    v16bf afr[4];
#pragma unroll
    for (int rt = 0; rt < 4; ++rt)
      afr[rt] = frag_a(&hs[(rt * 16 + l15) * HP + half * 8]);
#pragma unroll
    for (int ct = 0; ct < 8; ++ct) {
      const __bf16* wp = w2s + (wid * 128 + ct * 16 + l15) * W2P + half * 8;
      v16bf bfr = frag_b(wp);   // LDS (TDM-staged)
#pragma unroll
      for (int rt = 0; rt < 4; ++rt)
        acc[rt][ct] = __builtin_amdgcn_wmma_f32_16x16x32_bf16(
            false, afr[rt], false, bfr, (short)0, acc[rt][ct], false, false);
    }
  }

  // ---- epilogue: + b2, * combine weight, deterministic 2-way atomic combine
#pragma unroll
  for (int ct = 0; ct < 8; ++ct) {
    const int col = wid * 128 + ct * 16 + l15;
    const float b2v = b2[e * DM + col];
#pragma unroll
    for (int rt = 0; rt < 4; ++rt)
#pragma unroll
      for (int r = 0; r < 8; ++r) {
        const int row = rt * 16 + r + half * 8;
        if (row < rows) {
          const float v = wts[row] * (acc[rt][ct][r] + b2v);
          atomicAdd(&out[(size_t)toks[row] * DM + col], v);
        }
      }
  }
}

extern "C" void kernel_launch(void* const* d_in, const int* in_sizes, int n_in,
                              void* d_out, int out_size, void* d_ws, size_t ws_size,
                              hipStream_t stream) {
  const float* x  = (const float*)d_in[0];
  const float* Wg = (const float*)d_in[1];
  const float* bg = (const float*)d_in[2];
  const float* W1 = (const float*)d_in[3];
  const float* b1 = (const float*)d_in[4];
  const float* W2 = (const float*)d_in[5];
  const float* b2 = (const float*)d_in[6];
  float* out = (float*)d_out;

  const int T = in_sizes[0] / DM;  // 8192 tokens

  char* p = (char*)d_ws;
  auto carve = [&](size_t bytes) -> char* {
    char* r = p; p += (bytes + 255) & ~(size_t)255; return r;
  };
  __bf16* xb    = (__bf16*)carve((size_t)T * DM * sizeof(__bf16));
  __bf16* w1T   = (__bf16*)carve((size_t)NE * DF * DM * sizeof(__bf16));
  __bf16* w2T   = (__bf16*)carve((size_t)NE * DM * DF * sizeof(__bf16));
  int*    cnt   = (int*)carve(NE * sizeof(int));
  int*    list  = (int*)carve((size_t)NE * T * sizeof(int));
  float*  wlist = (float*)carve((size_t)NE * T * sizeof(float));

  (void)hipMemsetAsync(out, 0, (size_t)T * DM * sizeof(float), stream);
  (void)hipMemsetAsync(cnt, 0, NE * sizeof(int), stream);

  {
    const int n = T * DM;
    k_cvt_bf16<<<(n + 255) / 256, 256, 0, stream>>>(x, xb, n);
  }
  dim3 tb(32, 8);
  k_transpose_bf16<<<dim3(DF / 32, DM / 32, NE), tb, 0, stream>>>(W1, w1T, DM, DF);
  k_transpose_bf16<<<dim3(DM / 32, DF / 32, NE), tb, 0, stream>>>(W2, w2T, DF, DM);

  k_gate<<<(T + 7) / 8, 256, 0, stream>>>(x, Wg, bg, cnt, list, wlist, T);

  k_moe_ffn<<<dim3((T + TMT - 1) / TMT, NE), 256, SMEM_BYTES, stream>>>(
      xb, w1T, w2T, b1, b2, cnt, list, wlist, out, T);
}
